// SDPATransformerEncoderLayer_30829275251258
// MI455X (gfx1250) — matrix operand order repaired and checked by reading, not compiled, not run on hardware
//
#include <hip/hip_runtime.h>
#include <cstdint>
#include <cstddef>

// ---------------------------------------------------------------------------
// Types for CDNA5 WMMA (wave32): V_WMMA_F32_16X16X32_BF16
// ---------------------------------------------------------------------------
typedef __attribute__((ext_vector_type(16))) __bf16 v16bf;
typedef __attribute__((ext_vector_type(8)))  float  v8f;

struct FragU { uint32_t u[8]; };   // 8 VGPRs = 16 bf16 per lane

// ---------------------------------------------------------------------------
// Async global->LDS copy (GLOBAL_LOAD_ASYNC_TO_LDS_B128, ASYNCcnt) if the
// toolchain exposes the builtin; otherwise synchronous b128 copy fallback.
// Builtin signature (from hipcc diagnostic): arg0 = v4i32 in AS(1),
// arg1 = v4i32 in AS(3), then imm offset + imm cpol.
// ---------------------------------------------------------------------------
#if defined(__has_builtin)
#if __has_builtin(__builtin_amdgcn_global_load_async_to_lds_b128)
#define HAVE_ASYNC_B128 1
#endif
#endif

#ifdef HAVE_ASYNC_B128
typedef int i32x4 __attribute__((ext_vector_type(4)));
typedef __attribute__((address_space(1))) i32x4 g_i32x4;
typedef __attribute__((address_space(3))) i32x4 l_i32x4;
__device__ __forceinline__ void stage_b128(const void* g, void* l) {
  __builtin_amdgcn_global_load_async_to_lds_b128((g_i32x4*)g, (l_i32x4*)l, 0, 0);
}
__device__ __forceinline__ void stage_wait() {
  asm volatile("s_wait_asynccnt 0x0" ::: "memory");
}
#else
__device__ __forceinline__ void stage_b128(const void* g, void* l) {
  *(uint4*)l = *(const uint4*)g;
}
__device__ __forceinline__ void stage_wait() {}
#endif

__device__ __forceinline__ v8f v8f_zero() {
  v8f z;
#pragma unroll
  for (int i = 0; i < 8; ++i) z[i] = 0.0f;
  return z;
}

__device__ __forceinline__ uint16_t f2bf(float f) {
  uint32_t u = __float_as_uint(f);
  u += 0x7FFFu + ((u >> 16) & 1u);          // round-to-nearest-even
  return (uint16_t)(u >> 16);
}

__device__ __forceinline__ uint32_t pack_bf16(float lo, float hi) {
  return (uint32_t)f2bf(lo) | ((uint32_t)f2bf(hi) << 16);
}

// A-matrix fragment (16x32 bf16, MxK) from LDS (row-major, `stride` elems/row).
// Contiguous k order used here (B matches): lane m = lane&15; lanes0-15 hold K=0..15, lanes16-31 hold K=16..31.
__device__ __forceinline__ FragU load_afrag_lds(const uint16_t* base, int stride,
                                                int mbase, int kbase, int lane) {
  FragU f;
  const uint16_t* p = base + (mbase + (lane & 15)) * stride + kbase;
  int kb = (lane >> 4) << 4;
#pragma unroll
  for (int v = 0; v < 4; ++v) f.u[v]     = *(const uint32_t*)(p + kb + 2 * v);
#pragma unroll
  for (int v = 0; v < 4; ++v) f.u[4 + v] = *(const uint32_t*)(p + 8 + kb + 2 * v);
  return f;
}

// B-matrix fragment (32x16 bf16, KxN) from LDS stored TRANSPOSED as Bt[n][k].
// lanes0-15 hold K=0..15 (2/VGPR), lanes16-31 hold K=16..31.
__device__ __forceinline__ FragU load_bfrag_lds(const uint16_t* baseT, int stride,
                                                int nbase, int kbase, int lane) {
  FragU f;
  const uint16_t* p = baseT + (nbase + (lane & 15)) * stride + kbase + ((lane >> 4) << 4);
#pragma unroll
  for (int v = 0; v < 8; ++v) f.u[v] = *(const uint32_t*)(p + 2 * v);
  return f;
}

__device__ __forceinline__ v8f wmma32(const FragU& a, const FragU& b, v8f c) {
  union U { v16bf v; FragU f; } A, B;
  A.f = a; B.f = b;
  return __builtin_amdgcn_wmma_f32_16x16x32_bf16(
      /*neg_a=*/false, A.v, /*neg_b=*/false, B.v,
      /*c_mod=*/(short)0, c, /*reuse_a=*/false, /*reuse_b=*/false);
}

// ---------------------------------------------------------------------------
// fp32 -> bf16 conversion (elementwise, 4/thread)
// ---------------------------------------------------------------------------
__global__ __launch_bounds__(256) void cvt_k(const float* __restrict__ in,
                                             uint16_t* __restrict__ out, int n) {
  int i = (blockIdx.x * 256 + threadIdx.x) * 4;
  if (i + 3 < n) {
    float4 f = *(const float4*)(in + i);
    out[i + 0] = f2bf(f.x); out[i + 1] = f2bf(f.y);
    out[i + 2] = f2bf(f.z); out[i + 3] = f2bf(f.w);
  }
}

// ---------------------------------------------------------------------------
// Generic bf16 WMMA GEMM: C[M,N] = A[M,K] @ B[K,N] (+bias, +relu)
// block = 256 (8 waves), tile 128x128, K-step 32. Wave w: rows (w&3)*32..+32,
// cols (w>>2)*64..+64 -> 2x4 = 8 wmma accumulators. A tile staged with
// async-to-LDS b128; B tile transposed through VGPRs; next tiles prefetched.
// ---------------------------------------------------------------------------
#define LDT 40   // padded LDS row stride (bf16): 80B -> 16B aligned, conflict-free

template <bool HAS_BIAS, bool RELU, bool OUT_BF16>
__global__ __launch_bounds__(256) void gemm_bf16_k(
    const uint16_t* __restrict__ A, const uint16_t* __restrict__ B,
    const float* __restrict__ bias, float* __restrict__ Cf,
    uint16_t* __restrict__ Cb, int M, int N, int K) {
  __shared__ uint16_t As[128 * LDT];
  __shared__ uint16_t Bt[128 * LDT];

  const int tid = threadIdx.x;
  const int lane = tid & 31, wid = tid >> 5;
  const int rowBase = blockIdx.y * 128;
  const int colBase = blockIdx.x * 128;
  const int wm = (wid & 3) * 32;
  const int wn = (wid >> 2) * 64;

  v8f acc[2][4];
#pragma unroll
  for (int i = 0; i < 2; ++i)
#pragma unroll
    for (int j = 0; j < 4; ++j) acc[i][j] = v8f_zero();

  // Per-thread staging coordinates (fixed across K loop).
  const int ar0 = tid >> 2, aks = (tid & 3) * 8;            // A: idx, idx+256
  const int bk = tid >> 5, bn0 = (tid & 31) * 4;            // B: 4 slices

  for (int kc = 0; kc < K; kc += 32) {
    // Stage A tile (128x32) via async b128 copies into LDS.
#pragma unroll
    for (int l = 0; l < 2; ++l) {
      int r = ar0 + l * 64;
      const uint16_t* gp = A + (size_t)(rowBase + r) * K + kc + aks;
      stage_b128(gp, &As[r * LDT + aks]);
    }
    // Stage B tile (32x128) transposed -> Bt[n][k].
#pragma unroll
    for (int l = 0; l < 4; ++l) {
      int k = bk + l * 8;
      const uint16_t* gp = B + (size_t)(kc + k) * N + colBase + bn0;
      uint2 d = *(const uint2*)gp;
      Bt[(bn0 + 0) * LDT + k] = (uint16_t)(d.x & 0xffffu);
      Bt[(bn0 + 1) * LDT + k] = (uint16_t)(d.x >> 16);
      Bt[(bn0 + 2) * LDT + k] = (uint16_t)(d.y & 0xffffu);
      Bt[(bn0 + 3) * LDT + k] = (uint16_t)(d.y >> 16);
    }
    // Prefetch next K-step tiles while this tile is consumed.
    if (kc + 32 < K) {
      __builtin_prefetch(A + (size_t)(rowBase + ar0) * K + kc + 32 + aks, 0, 3);
      __builtin_prefetch(B + (size_t)(kc + 32 + bk) * N + colBase + bn0, 0, 3);
    }
    stage_wait();
    __syncthreads();

    FragU af0 = load_afrag_lds(As, LDT, wm, 0, lane);
    FragU af1 = load_afrag_lds(As, LDT, wm + 16, 0, lane);
#pragma unroll
    for (int nt = 0; nt < 4; ++nt) {
      FragU bfr = load_bfrag_lds(Bt, LDT, wn + nt * 16, 0, lane);
      acc[0][nt] = wmma32(af0, bfr, acc[0][nt]);
      acc[1][nt] = wmma32(af1, bfr, acc[1][nt]);
    }
    __syncthreads();
  }

  // Epilogue. C layout: VGPR r -> row r (+8 for lanes 16..31), col = lane&15.
#pragma unroll
  for (int mt = 0; mt < 2; ++mt)
#pragma unroll
    for (int nt = 0; nt < 4; ++nt) {
      const int col = colBase + wn + nt * 16 + (lane & 15);
      float bv = 0.0f;
      if constexpr (HAS_BIAS) bv = bias[col];
      const int row0 = rowBase + wm + mt * 16 + ((lane >> 4) << 3);
      if constexpr (OUT_BF16) {
        uint16_t* p = Cb + (size_t)row0 * N + col;
#pragma unroll
        for (int r = 0; r < 8; ++r) {
          float vv = acc[mt][nt][r] + bv;
          if constexpr (RELU) vv = fmaxf(vv, 0.0f);
          *p = f2bf(vv);
          p += N;
        }
      } else {
        float* p = Cf + (size_t)row0 * N + col;
#pragma unroll
        for (int r = 0; r < 8; ++r) {
          float vv = acc[mt][nt][r] + bv;
          if constexpr (RELU) vv = fmaxf(vv, 0.0f);
          *p = vv;
          p += N;
        }
      }
    }
}

// ---------------------------------------------------------------------------
// Repack qkv fp32 [B*S, 3*D] -> q,k,v bf16 [B,H,S,HD], q pre-scaled by 1/sqrt(64)
// ---------------------------------------------------------------------------
__global__ __launch_bounds__(256) void repack_qkv_k(
    const float* __restrict__ qkv, uint16_t* __restrict__ qo,
    uint16_t* __restrict__ ko, uint16_t* __restrict__ vo) {
  size_t idx = (size_t)blockIdx.x * 256 + threadIdx.x;   // B*S*3D elems
  int col = (int)(idx % 3072);
  size_t rowS = idx / 3072;
  int part = col >> 10, rem = col & 1023;
  int h = rem >> 6, hd = rem & 63;
  int b = (int)(rowS >> 11), s = (int)(rowS & 2047);
  float val = qkv[idx];
  size_t dst = (((size_t)b * 16 + h) * 2048 + s) * 64 + hd;
  if (part == 0)      qo[dst] = f2bf(val * 0.125f);
  else if (part == 1) ko[dst] = f2bf(val);
  else                vo[dst] = f2bf(val);
}

// ---------------------------------------------------------------------------
// Attention: one workgroup per (b, h, 32-query tile).
// Full 32x2048 fp32 score panel resident in LDS (~257KB of the 320KB/WGP),
// exact softmax in LDS, then ctx = P @ V with on-the-fly fp32->bf16 A-frags.
// ---------------------------------------------------------------------------
#define SSTR 2052   // padded fp32 score row stride (4-bank step, conflict-free)
#define QLD  72     // padded q tile row stride (bf16)
#define VLD  40     // padded Vt row stride (bf16)

__global__ __launch_bounds__(256) void attn_k(
    const uint16_t* __restrict__ qg, const uint16_t* __restrict__ kg,
    const uint16_t* __restrict__ vg, uint16_t* __restrict__ ctx) {
  extern __shared__ char smem[];
  float*    sc = (float*)smem;                                     // 32 x 2052 fp32
  uint16_t* qs = (uint16_t*)(smem + 32 * SSTR * 4);                // 32 x 72 bf16
  uint16_t* Vt = (uint16_t*)(smem + 32 * SSTR * 4 + 32 * QLD * 2); // 64 x 40 bf16

  const int tid = threadIdx.x, lane = tid & 31, wid = tid >> 5;
  const int qt = blockIdx.x, h = blockIdx.y, b = blockIdx.z;
  const size_t hb = ((size_t)b * 16 + h) * 2048 * 64;
  const uint16_t* qp = qg + hb;
  const uint16_t* kp = kg + hb;
  const uint16_t* vp = vg + hb;
  const int q0 = qt * 32;

  // Phase 0: stage q tile (already pre-scaled bf16) via async b128.
  {
    int r = tid >> 3, c0 = (tid & 7) * 8;
    stage_b128(&qp[(size_t)(q0 + r) * 64 + c0], &qs[r * QLD + c0]);
    stage_wait();
  }
  __syncthreads();

  // Phase 1: scores = q @ k^T  (32 x 2048), K-dim = HD = 64 -> 2 chained wmma.
  FragU aq[2][2];
#pragma unroll
  for (int mt = 0; mt < 2; ++mt)
#pragma unroll
    for (int kk = 0; kk < 2; ++kk)
      aq[mt][kk] = load_afrag_lds(qs, QLD, mt * 16, kk * 32, lane);

  for (int it = 0; it < 16; ++it) {
    int ncol = (wid + it * 8) * 16;   // 16-key column tile owned by this wave
    // B fragments straight from global k (row = key, contiguous HD pairs).
    FragU bk0, bk1;
    {
      const uint16_t* p = kp + (size_t)(ncol + (lane & 15)) * 64 + ((lane >> 4) << 4);
#pragma unroll
      for (int v = 0; v < 8; ++v) bk0.u[v] = *(const uint32_t*)(p + 2 * v);
#pragma unroll
      for (int v = 0; v < 8; ++v) bk1.u[v] = *(const uint32_t*)(p + 32 + 2 * v);
    }
#pragma unroll
    for (int mt = 0; mt < 2; ++mt) {
      v8f c = v8f_zero();
      c = wmma32(aq[mt][0], bk0, c);
      c = wmma32(aq[mt][1], bk1, c);
      int colw = ncol + (lane & 15);
#pragma unroll
      for (int r = 0; r < 8; ++r) {
        int row = mt * 16 + r + ((lane >> 4) << 3);
        sc[row * SSTR + colw] = c[r];
      }
    }
  }
  __syncthreads();

  // Phase 2: exact softmax over each 2048-wide row (8 lanes per row).
  {
    int row = tid >> 3, l8 = tid & 7;
    float* srow = sc + row * SSTR;
    float mx = -1e30f;
    for (int c = l8; c < 2048; c += 8) mx = fmaxf(mx, srow[c]);
    mx = fmaxf(mx, __shfl_xor(mx, 1, 8));
    mx = fmaxf(mx, __shfl_xor(mx, 2, 8));
    mx = fmaxf(mx, __shfl_xor(mx, 4, 8));
    float sum = 0.0f;
    for (int c = l8; c < 2048; c += 8) {
      float e = __expf(srow[c] - mx);
      srow[c] = e;
      sum += e;
    }
    sum += __shfl_xor(sum, 1, 8);
    sum += __shfl_xor(sum, 2, 8);
    sum += __shfl_xor(sum, 4, 8);
    float inv = 1.0f / sum;
    for (int c = l8; c < 2048; c += 8) srow[c] *= inv;
  }
  __syncthreads();

  // Phase 3: ctx = P @ V   (32 x 64), wave owns one 16x16 tile, K = 2048.
  const int mt = wid >> 2, nt = wid & 3;
  v8f acc = v8f_zero();
  for (int kc = 0; kc < 64; ++kc) {
    __syncthreads();
    // Stage V chunk (32 keys x 64 hd) transposed -> Vt[hd][key].
#pragma unroll
    for (int l = 0; l < 2; ++l) {
      int idx = tid + l * 256;                // 0..511
      int key = idx >> 4, c0 = (idx & 15) * 4;
      uint2 d = *(const uint2*)&vp[(size_t)(kc * 32 + key) * 64 + c0];
      Vt[(c0 + 0) * VLD + key] = (uint16_t)(d.x & 0xffffu);
      Vt[(c0 + 1) * VLD + key] = (uint16_t)(d.x >> 16);
      Vt[(c0 + 2) * VLD + key] = (uint16_t)(d.y & 0xffffu);
      Vt[(c0 + 3) * VLD + key] = (uint16_t)(d.y >> 16);
    }
    __syncthreads();
    // A-fragment: convert normalized fp32 probabilities -> packed bf16 pairs.
    FragU ap;
    {
      const float* sr = sc + (mt * 16 + (lane & 15)) * SSTR + kc * 32;
      int kb = (lane >> 4) << 4;
#pragma unroll
      for (int v = 0; v < 4; ++v)
        ap.u[v] = pack_bf16(sr[kb + 2 * v], sr[kb + 2 * v + 1]);
#pragma unroll
      for (int v = 0; v < 4; ++v)
        ap.u[4 + v] = pack_bf16(sr[8 + kb + 2 * v], sr[8 + kb + 2 * v + 1]);
    }
    FragU bvf = load_bfrag_lds(Vt, VLD, nt * 16, 0, lane);
    acc = wmma32(ap, bvf, acc);
  }

  // Store ctx tile as bf16 into [B*S, D] with col = h*64 + hd.
  {
    int col = h * 64 + nt * 16 + (lane & 15);
    uint16_t* p = ctx + (size_t)((size_t)b * 2048 + q0 + mt * 16 + ((lane >> 4) << 3)) * 1024 + col;
#pragma unroll
    for (int r = 0; r < 8; ++r) {
      *p = f2bf(acc[r]);
      p += 1024;
    }
  }
}

// ---------------------------------------------------------------------------
// Fused residual-add + LayerNorm (block per row, D = 1024).
// ---------------------------------------------------------------------------
template <bool OUT_BF16_TOO>
__global__ __launch_bounds__(256) void add_ln_k(
    const float* __restrict__ a, const float* __restrict__ rsd,
    const float* __restrict__ g, const float* __restrict__ be,
    float* __restrict__ outf, uint16_t* __restrict__ outb) {
  __shared__ float red[16];
  const int row = blockIdx.x, tid = threadIdx.x;
  const int lane = tid & 31, wid = tid >> 5;
  const float* pa = a + (size_t)row * 1024;
  const float* pr = rsd + (size_t)row * 1024;

  float x[4], s = 0.0f, s2 = 0.0f;
#pragma unroll
  for (int j = 0; j < 4; ++j) {
    int i = tid + j * 256;
    x[j] = pa[i] + pr[i];
    s += x[j]; s2 += x[j] * x[j];
  }
#pragma unroll
  for (int o = 1; o < 32; o <<= 1) {
    s  += __shfl_xor(s, o, 32);
    s2 += __shfl_xor(s2, o, 32);
  }
  if (lane == 0) { red[wid] = s; red[8 + wid] = s2; }
  __syncthreads();
  if (wid == 0) {
    float a0 = (lane < 8) ? red[lane] : 0.0f;
    float b0 = (lane < 8) ? red[8 + lane] : 0.0f;
#pragma unroll
    for (int o = 1; o < 8; o <<= 1) {
      a0 += __shfl_xor(a0, o, 32);
      b0 += __shfl_xor(b0, o, 32);
    }
    if (lane == 0) { red[0] = a0; red[8] = b0; }
  }
  __syncthreads();
  float mu = red[0] * (1.0f / 1024.0f);
  float var = red[8] * (1.0f / 1024.0f) - mu * mu;
  float rstd = rsqrtf(var + 1e-5f);
#pragma unroll
  for (int j = 0; j < 4; ++j) {
    int i = tid + j * 256;
    float y = (x[j] - mu) * rstd * g[i] + be[i];
    outf[(size_t)row * 1024 + i] = y;
    if constexpr (OUT_BF16_TOO) outb[(size_t)row * 1024 + i] = f2bf(y);
  }
}

// ---------------------------------------------------------------------------
// Host orchestration
// ---------------------------------------------------------------------------
extern "C" void kernel_launch(void* const* d_in, const int* in_sizes, int n_in,
                              void* d_out, int out_size, void* d_ws, size_t ws_size,
                              hipStream_t stream) {
  (void)in_sizes; (void)n_in; (void)out_size; (void)ws_size;
  const float* src  = (const float*)d_in[0];
  const float* Wqkv = (const float*)d_in[1];
  const float* Wout = (const float*)d_in[2];
  const float* W1   = (const float*)d_in[3];
  const float* W2   = (const float*)d_in[4];
  const float* b1   = (const float*)d_in[5];
  const float* b2   = (const float*)d_in[6];
  const float* g1   = (const float*)d_in[7];
  const float* be1  = (const float*)d_in[8];
  const float* g2   = (const float*)d_in[9];
  const float* be2  = (const float*)d_in[10];
  float* out = (float*)d_out;

  char* ws = (char*)d_ws;
  size_t off = 0;
  auto alloc = [&](size_t bytes) -> void* {
    void* p = ws + off;
    off = (off + bytes + 255) & ~(size_t)255;
    return p;
  };

  const size_t BS = 8192;            // B*S
  uint16_t* srcb  = (uint16_t*)alloc(BS * 1024 * 2);
  uint16_t* wqkvb = (uint16_t*)alloc(1024 * 3072 * 2);
  uint16_t* woutb = (uint16_t*)alloc(1024 * 1024 * 2);
  uint16_t* w1b   = (uint16_t*)alloc(1024 * 4096 * 2);
  uint16_t* w2b   = (uint16_t*)alloc(4096 * 1024 * 2);
  float*    qkvf  = (float*)   alloc(BS * 3072 * 4);   // reused below
  uint16_t* qb    = (uint16_t*)alloc(BS * 1024 * 2);
  uint16_t* kb    = (uint16_t*)alloc(BS * 1024 * 2);
  uint16_t* vb    = (uint16_t*)alloc(BS * 1024 * 2);
  uint16_t* ctxb  = (uint16_t*)alloc(BS * 1024 * 2);
  float*    xf    = (float*)   alloc(BS * 1024 * 4);
  uint16_t* xb    = (uint16_t*)alloc(BS * 1024 * 2);
  uint16_t* hb    = (uint16_t*)alloc(BS * 4096 * 2);
  float* attn_out = qkvf;                        // qkv fp32 dead after repack
  float* ff       = qkvf + BS * 1024;            // still inside qkvf region

  dim3 blk(256);

  // fp32 -> bf16 operands
  cvt_k<<<8192, blk, 0, stream>>>(src,  srcb,  8192 * 1024);
  cvt_k<<<3072, blk, 0, stream>>>(Wqkv, wqkvb, 1024 * 3072);
  cvt_k<<<1024, blk, 0, stream>>>(Wout, woutb, 1024 * 1024);
  cvt_k<<<4096, blk, 0, stream>>>(W1,   w1b,   1024 * 4096);
  cvt_k<<<4096, blk, 0, stream>>>(W2,   w2b,   4096 * 1024);

  // qkv = src @ Wqkv  [8192,3072]
  gemm_bf16_k<false, false, false><<<dim3(24, 64), blk, 0, stream>>>(
      srcb, wqkvb, nullptr, qkvf, nullptr, 8192, 3072, 1024);
  // split heads, fold 1/sqrt(HD) into q
  repack_qkv_k<<<98304, blk, 0, stream>>>(qkvf, qb, kb, vb);

  // attention (~264KB dynamic LDS per workgroup)
  const int attn_lds = 32 * SSTR * 4 + 32 * QLD * 2 + 64 * VLD * 2;
  (void)hipFuncSetAttribute(reinterpret_cast<const void*>(attn_k),
                            hipFuncAttributeMaxDynamicSharedMemorySize, attn_lds);
  attn_k<<<dim3(64, 16, 4), blk, attn_lds, stream>>>(qb, kb, vb, ctxb);

  // attn_out = ctx @ Wout
  gemm_bf16_k<false, false, false><<<dim3(8, 64), blk, 0, stream>>>(
      ctxb, woutb, nullptr, attn_out, nullptr, 8192, 1024, 1024);
  // x = LN(src + attn_out)
  add_ln_k<true><<<8192, blk, 0, stream>>>(src, attn_out, g1, be1, xf, xb);

  // h = relu(x @ W1 + b1) -> bf16
  gemm_bf16_k<true, true, true><<<dim3(32, 64), blk, 0, stream>>>(
      xb, w1b, b1, nullptr, hb, 8192, 4096, 1024);
  // ff = h @ W2 + b2
  gemm_bf16_k<true, false, false><<<dim3(8, 64), blk, 0, stream>>>(
      hb, w2b, b2, ff, nullptr, 8192, 1024, 4096);
  // out = LN(x + ff)
  add_ln_k<false><<<8192, blk, 0, stream>>>(xf, ff, g2, be2, out, nullptr);
}
